// ResidualAttention_3264175145036
// MI455X (gfx1250) — compile-verified, matrix operand order checked by reading, not run on hardware
//
#include <hip/hip_runtime.h>
#include <hip/hip_bf16.h>

// ---------------------------------------------------------------------------
// Problem constants (B,N,M,DQ,DK,DV,H) = (8,1024,1024,512,512,512,8), DS=64
// ---------------------------------------------------------------------------
#define BB   8
#define NN   1024
#define MM   1024
#define DD   512          // DQ=DK=DV
#define HH   8
#define DSS  64
#define ROWS (BB * NN)    // 8192

typedef __attribute__((ext_vector_type(8)))  float        v8f;
typedef __attribute__((ext_vector_type(16))) __bf16       v16bf;
typedef __attribute__((ext_vector_type(4)))  unsigned int u32x4;
typedef __attribute__((ext_vector_type(4)))  unsigned int vu4;
typedef __attribute__((ext_vector_type(8)))  int          vi8;
typedef __attribute__((ext_vector_type(4)))  int          vi4;

struct BF16Frag { u32x4 lo, hi; };

__device__ __forceinline__ unsigned short f2bf(float f) {
  unsigned int u = __builtin_bit_cast(unsigned int, f);
  u += 0x7FFFu + ((u >> 16) & 1u);               // round-to-nearest-even
  return (unsigned short)(u >> 16);
}

// Load one 16x32 bf16 A/B fragment: 8 bf16 at +kb, 8 bf16 at +kb+16
__device__ __forceinline__ v16bf load_frag(const unsigned short* p, int kb) {
  BF16Frag t;
  t.lo = *reinterpret_cast<const u32x4*>(p + kb);
  t.hi = *reinterpret_cast<const u32x4*>(p + kb + 16);
  return __builtin_bit_cast(v16bf, t);
}

__device__ __forceinline__ v8f wmma_bf16(v16bf a, v16bf b, v8f c) {
  return __builtin_amdgcn_wmma_f32_16x16x32_bf16(
      /*neg_a=*/false, a, /*neg_b=*/false, b,
      /*c_mod=*/(short)0, c, /*reuse_a=*/false, /*reuse_b=*/false);
}

// ---------------------------------------------------------------------------
// Tensor Data Mover: DMA one [tile_rows x 32] bf16 tile (row pitch 512 elems)
// into LDS with 16B padding every 64B row (-> 40-short LDS pitch).
// D# layout per CDNA5 ISA ch.8: group0 = {flags, lds_addr, global_addr, type},
// group1 = {data_size/pad, tensor dims, tile dims, dim0 stride}.
// This toolchain exposes the 6-arg builtin (g0, g1, g2, g3, g4, cpol).
// ---------------------------------------------------------------------------
#define TK  32
#define BT  128
#define LDR 40    // lds row stride in shorts: 32 data + 8 pad (16B), 16B-aligned

__device__ __forceinline__ void tdm_load_tile(const void* gsrc, unsigned lds_off,
                                              unsigned tile_rows, unsigned tensor_rows) {
  unsigned long long ga = (unsigned long long)gsrc;
  vu4 g0;
  g0[0] = 1u;                                            // count=1, is_restore=0
  g0[1] = lds_off;                                       // lds_addr [63:32]
  g0[2] = (unsigned)ga;                                  // global_addr [95:64]
  g0[3] = (unsigned)((ga >> 32) & 0x01FFFFFFu)           // global_addr [120:96]
        | (2u << 30);                                    // type=2 (image)
  vi8 g1;
  g1[0] = (1 << 16)                                      // data_size = 2 bytes
        | (1 << 20)                                      // pad_enable
        | (3 << 22)                                      // pad_interval: 16 DWORDs (64B)
        | (3 << 25);                                     // pad_amount: 4 DWORDs (16B)
  g1[1] = (int)(((unsigned)DD & 0xFFFFu) << 16);         // tensor_dim0 = 512 (lo16)
  g1[2] = (int)((tensor_rows & 0xFFFFu) << 16);          // dim0 hi=0 | tensor_dim1 lo16
  g1[3] = (int)((tensor_rows >> 16) & 0xFFFFu)           // tensor_dim1 hi16
        | (TK << 16);                                    // tile_dim0 = 32
  g1[4] = (int)(tile_rows & 0xFFFFu);                    // tile_dim1 | tile_dim2=0
  g1[5] = DD;                                            // tensor_dim0_stride = 512
  g1[6] = 0;
  g1[7] = 0;
  vi4 z4 = (vi4)0;                                       // groups 2/3 unused (2D tensor)
  vi8 z8 = (vi8)0;
  __builtin_amdgcn_tensor_load_to_lds(g0, g1, z4, z4, z8, /*cpol=*/0);
}

// ---------------------------------------------------------------------------
// fp32 -> bf16 conversion kernels
// ---------------------------------------------------------------------------
__global__ __launch_bounds__(256)
void cvt_act_kernel(const float* __restrict__ in, const unsigned char* __restrict__ mask,
                    unsigned short* __restrict__ out) {
  size_t i = (size_t)blockIdx.x * 256 + threadIdx.x;       // ROWS*DD elems
  int row = (int)(i >> 9);
  float v = mask[row] ? 0.0f : in[i];
  out[i] = f2bf(v);
}

// W[K][N] fp32 -> Wt[N][K] bf16 (so B-fragment columns are contiguous)
__global__ __launch_bounds__(256)
void cvt_wt_kernel(const float* __restrict__ w, unsigned short* __restrict__ wt) {
  int i = blockIdx.x * 256 + threadIdx.x;                  // DD*DD elems
  int r = i >> 9, c = i & 511;
  wt[c * DD + r] = f2bf(w[r * DD + c]);
}

// v_rowmajor [B,M,H*DS] bf16 -> Vt [B,H,DS,M] bf16
__global__ __launch_bounds__(256)
void transpose_v_kernel(const unsigned short* __restrict__ vrm,
                        unsigned short* __restrict__ vt) {
  size_t i = (size_t)blockIdx.x * 256 + threadIdx.x;       // B*M*DD
  int b = (int)(i >> 19);
  int m = (int)(i >> 9) & (MM - 1);
  int hd = (int)i & (DD - 1);
  int h = hd >> 6, d = hd & 63;
  vt[(((size_t)(b * HH + h) * DSS + d) * MM) + m] = vrm[i];
}

// ---------------------------------------------------------------------------
// Generic bf16 WMMA GEMM: C[ROWS x 512] = A[ROWS x 512] @ Bt[512 x 512]^T
// FLAGS bit0: output f32 (else bf16); bit1: +bias; bit2: ReLU
// Block: 128 threads (4 waves), tile 128x128, each wave 64x64.
// K tiles are DMA'd into double-buffered LDS by the Tensor Data Mover
// (wave 0 issues, TENSORcnt-tracked) while all waves run WMMA.
// ---------------------------------------------------------------------------
template<int FLAGS>
__global__ __launch_bounds__(128)
void gemm_bf16_kernel(const unsigned short* __restrict__ A,
                      const unsigned short* __restrict__ Bt,
                      const float* __restrict__ bias,
                      void* __restrict__ C) {
  __shared__ unsigned short sA[2][BT * LDR];
  __shared__ unsigned short sB[2][BT * LDR];

  const int row0 = blockIdx.y * BT;
  const int col0 = blockIdx.x * BT;
  const int tid  = threadIdx.x;
  const int lane = tid & 31;
  const int wave = tid >> 5;
  const int wm   = (wave >> 1) * 64;
  const int wn   = (wave & 1) * 64;
  const int lr   = lane & 15;
  const int lh   = lane >> 4;
  const int kb   = lh ? 8 : 0;

  // LDS byte offsets (flat shared VA carries LDS offset in addr[31:0])
  const unsigned ldsA0 = (unsigned)(unsigned long long)(&sA[0][0]);
  const unsigned ldsA1 = (unsigned)(unsigned long long)(&sA[1][0]);
  const unsigned ldsB0 = (unsigned)(unsigned long long)(&sB[0][0]);
  const unsigned ldsB1 = (unsigned)(unsigned long long)(&sB[1][0]);

  v8f acc[4][4];
#pragma unroll
  for (int i = 0; i < 4; ++i)
#pragma unroll
    for (int j = 0; j < 4; ++j) acc[i][j] = (v8f)0.0f;

  const int nk = DD / TK;   // 16 K-steps

  // prologue: DMA first K tile pair
  if (wave == 0) {
    tdm_load_tile(A + (size_t)row0 * DD, ldsA0, BT, ROWS);
    tdm_load_tile(Bt + (size_t)col0 * DD, ldsB0, BT, DD);
    __builtin_amdgcn_s_wait_tensorcnt(0);
  }
  __syncthreads();

  for (int kc = 0; kc < nk; ++kc) {
    const int cur = kc & 1;
    const unsigned short* curA = sA[cur];
    const unsigned short* curB = sB[cur];

    // kick off DMA of the next tile pair into the other buffer
    if (wave == 0 && kc + 1 < nk) {
      int k1 = (kc + 1) * TK;
      tdm_load_tile(A + (size_t)row0 * DD + k1, cur ? ldsA0 : ldsA1, BT, ROWS);
      tdm_load_tile(Bt + (size_t)col0 * DD + k1, cur ? ldsB0 : ldsB1, BT, DD);
    }

    v16bf af[4], bfr[4];
#pragma unroll
    for (int i = 0; i < 4; ++i)
      af[i] = load_frag(curA + (wm + i * 16 + lr) * LDR, kb);
#pragma unroll
    for (int j = 0; j < 4; ++j)
      bfr[j] = load_frag(curB + (wn + j * 16 + lr) * LDR, kb);

#pragma unroll
    for (int i = 0; i < 4; ++i)
#pragma unroll
      for (int j = 0; j < 4; ++j)
        acc[i][j] = wmma_bf16(af[i], bfr[j], acc[i][j]);

    if (wave == 0 && kc + 1 < nk) __builtin_amdgcn_s_wait_tensorcnt(0);
    __syncthreads();
  }

  // epilogue: D layout is VGPR r -> row (r + 8*lh), col = lane&15
#pragma unroll
  for (int i = 0; i < 4; ++i) {
#pragma unroll
    for (int j = 0; j < 4; ++j) {
#pragma unroll
      for (int r = 0; r < 8; ++r) {
        int grow = row0 + wm + i * 16 + r + 8 * lh;
        int gcol = col0 + wn + j * 16 + lr;
        float v = acc[i][j][r];
        if constexpr (FLAGS & 2) v += bias[gcol];
        if constexpr (FLAGS & 4) v = fmaxf(v, 0.0f);
        size_t idx = (size_t)grow * DD + gcol;
        if constexpr (FLAGS & 1) reinterpret_cast<float*>(C)[idx] = v;
        else reinterpret_cast<unsigned short*>(C)[idx] = f2bf(v);
      }
    }
  }
}

// ---------------------------------------------------------------------------
// Attention: one block (2 waves) per (b, h, 32-row tile of N).
// Full 32x1024 score block lives in LDS (128 KB of the WGP's 320 KB);
// probabilities are re-packed bf16 in place for the PV WMMA pass.
// q,k: bf16 [B, N/M, H, DS]; Vt: bf16 [B, H, DS, M]; att: bf16 [B, N, DD]
// ---------------------------------------------------------------------------
__global__ __launch_bounds__(64)
void attn_kernel(const unsigned short* __restrict__ q,
                 const unsigned short* __restrict__ k,
                 const unsigned short* __restrict__ vt,
                 const unsigned char* __restrict__ xmask,
                 const unsigned char* __restrict__ ymask,
                 unsigned short* __restrict__ att) {
  extern __shared__ char smem[];
  float* S = reinterpret_cast<float*>(smem);              // [32][1024] fp32
  unsigned short* Ps = reinterpret_cast<unsigned short*>(smem);  // bf16 alias, stride 2048

  const int bid = blockIdx.x;
  const int nt  = bid & 31;
  const int h   = (bid >> 5) & 7;
  const int b   = bid >> 8;
  const int n0  = nt * 32;

  const int tid  = threadIdx.x;
  const int lane = tid & 31;
  const int w    = tid >> 5;
  const int lr   = lane & 15;
  const int lh   = lane >> 4;
  const int kb   = lh ? 8 : 0;
  const float scale = 0.04419417382415922f;               // 1/sqrt(512)

  // ---- phase 1: S = (q k^T) * scale, key-mask applied -------------------
  const int qrow = n0 + w * 16 + lr;                      // A-frag row (M = lane&15)
  const unsigned short* qp = q + ((size_t)(b * NN + qrow) * HH + h) * DSS;
  v16bf qa0 = load_frag(qp, kb);                          // d = 0..31
  v16bf qa1 = load_frag(qp + 32, kb);                     // d = 32..63

  for (int mt = 0; mt < MM / 16; ++mt) {
    int mcol = mt * 16 + lr;                              // B-frag column
    const unsigned short* kp = k + ((size_t)(b * MM + mcol) * HH + h) * DSS;
    v16bf kb0 = load_frag(kp, kb);
    v16bf kb1 = load_frag(kp + 32, kb);
    v8f s = (v8f)0.0f;
    s = wmma_bf16(qa0, kb0, s);
    s = wmma_bf16(qa1, kb1, s);
    bool ym = ymask[b * MM + mcol] != 0;
#pragma unroll
    for (int r = 0; r < 8; ++r) {
      int lrow = w * 16 + r + 8 * lh;
      S[lrow * MM + mcol] = ym ? -1e38f : s[r] * scale;
    }
  }
  __syncthreads();

  // ---- phase 2: rowwise softmax; write bf16 P in place over S -----------
  if (tid < 32) {
    int row = tid;
    float* Sr = S + (size_t)row * MM;
    unsigned short* Pr = Ps + (size_t)row * 2 * MM;       // same bytes as Sr
    float mx = -3.4e38f;
    for (int m = 0; m < MM; ++m) mx = fmaxf(mx, Sr[m]);
    float sum = 0.0f;
    for (int m = 0; m < MM; ++m) {
      float e = __expf(Sr[m] - mx);                       // read float m ...
      sum += e;
      Pr[m] = f2bf(e);                                    // ... write short m (float m/2: done)
    }
    float inv = 1.0f / sum;
    for (int m = 0; m < MM; ++m) {
      float p = (float)__builtin_bit_cast(__bf16, Pr[m]);
      Pr[m] = f2bf(p * inv);
    }
  }
  __syncthreads();

  // ---- phase 3: O = P @ V ----------------------------------------------
  v8f o[4];
#pragma unroll
  for (int dt = 0; dt < 4; ++dt) o[dt] = (v8f)0.0f;

  const unsigned short* Pw = Ps + (size_t)(w * 16 + lr) * 2 * MM;  // A-frag row base
  for (int mc = 0; mc < MM / 32; ++mc) {
    v16bf pa = load_frag(Pw + mc * 32, kb);
#pragma unroll
    for (int dt = 0; dt < 4; ++dt) {
      int d = dt * 16 + lr;                               // B-frag column
      const unsigned short* vp =
          vt + (((size_t)(b * HH + h) * DSS + d) * MM) + mc * 32;
      v16bf vb = load_frag(vp, kb);
      o[dt] = wmma_bf16(pa, vb, o[dt]);
    }
  }

  // ---- write att (bf16), query-mask rows to zero ------------------------
#pragma unroll
  for (int dt = 0; dt < 4; ++dt) {
#pragma unroll
    for (int r = 0; r < 8; ++r) {
      int n = n0 + w * 16 + r + 8 * lh;
      int d = dt * 16 + lr;
      float v = (xmask[b * NN + n] != 0) ? 0.0f : o[dt][r];
      att[(size_t)(b * NN + n) * DD + h * DSS + d] = f2bf(v);
    }
  }
}

// ---------------------------------------------------------------------------
// Residual + LayerNorm epilogues. One wave (32 lanes) per 512-wide row.
// FINAL=0: t = base + add; LN(g,be) -> outf (fp32) and outb (bf16)
// FINAL=1: t = base + add; LN(g,be); query-mask -> outf (fp32, = d_out)
// ---------------------------------------------------------------------------
template<int FINAL>
__global__ __launch_bounds__(256)
void ln_res_kernel(const float* __restrict__ base, const float* __restrict__ add,
                   const float* __restrict__ g, const float* __restrict__ be,
                   const unsigned char* __restrict__ xmask,
                   float* __restrict__ outf, unsigned short* __restrict__ outb) {
  int row  = blockIdx.x * 8 + (threadIdx.x >> 5);
  int lane = threadIdx.x & 31;
  const float* a = base + (size_t)row * DD;
  const float* c = add + (size_t)row * DD;

  float v[16];
  float s = 0.0f;
#pragma unroll
  for (int i = 0; i < 16; ++i) {
    int col = lane + i * 32;
    v[i] = a[col] + c[col];
    s += v[i];
  }
#pragma unroll
  for (int off = 16; off > 0; off >>= 1) s += __shfl_xor(s, off, 32);
  float mu = s * (1.0f / DD);

  float s2 = 0.0f;
#pragma unroll
  for (int i = 0; i < 16; ++i) { float d = v[i] - mu; s2 += d * d; }
#pragma unroll
  for (int off = 16; off > 0; off >>= 1) s2 += __shfl_xor(s2, off, 32);
  float rinv = rsqrtf(s2 * (1.0f / DD) + 1e-5f);

  bool xm = (FINAL == 1) && (xmask[row] != 0);
#pragma unroll
  for (int i = 0; i < 16; ++i) {
    int col = lane + i * 32;
    float o = (v[i] - mu) * rinv * g[col] + be[col];
    if constexpr (FINAL == 1) {
      outf[(size_t)row * DD + col] = xm ? 0.0f : o;
    } else {
      outf[(size_t)row * DD + col] = o;
      outb[(size_t)row * DD + col] = f2bf(o);
    }
  }
}

// ---------------------------------------------------------------------------
// Host orchestration
// ---------------------------------------------------------------------------
extern "C" void kernel_launch(void* const* d_in, const int* in_sizes, int n_in,
                              void* d_out, int out_size, void* d_ws, size_t ws_size,
                              hipStream_t stream) {
  (void)in_sizes; (void)n_in; (void)out_size; (void)ws_size;

  const float* x  = (const float*)d_in[0];
  const float* y  = (const float*)d_in[1];
  const unsigned char* xm = (const unsigned char*)d_in[2];
  const unsigned char* ym = (const unsigned char*)d_in[3];
  const float* Wq = (const float*)d_in[4];
  const float* Wk = (const float*)d_in[5];
  const float* Wv = (const float*)d_in[6];
  const float* Wo = (const float*)d_in[7];
  const float* W1 = (const float*)d_in[8];
  const float* b1 = (const float*)d_in[9];
  const float* W2 = (const float*)d_in[10];
  const float* b2 = (const float*)d_in[11];
  const float* g1 = (const float*)d_in[12];
  const float* be1 = (const float*)d_in[13];
  const float* g2 = (const float*)d_in[14];
  const float* be2 = (const float*)d_in[15];
  float* out = (float*)d_out;

  // workspace carve-up
  char* ws = (char*)d_ws;
  size_t off = 0;
  auto alloc = [&](size_t bytes) {
    char* p = ws + off;
    off = (off + bytes + 255) & ~(size_t)255;
    return p;
  };
  const size_t actB  = (size_t)ROWS * DD * sizeof(unsigned short);  // 8 MB
  const size_t actF  = (size_t)ROWS * DD * sizeof(float);           // 16 MB
  const size_t wtB   = (size_t)DD * DD * sizeof(unsigned short);    // 512 KB

  unsigned short* xb   = (unsigned short*)alloc(actB);
  unsigned short* yb   = (unsigned short*)alloc(actB);
  unsigned short* Wqt  = (unsigned short*)alloc(wtB);
  unsigned short* Wkt  = (unsigned short*)alloc(wtB);
  unsigned short* Wvt  = (unsigned short*)alloc(wtB);
  unsigned short* Wot  = (unsigned short*)alloc(wtB);
  unsigned short* W1t  = (unsigned short*)alloc(wtB);
  unsigned short* W2t  = (unsigned short*)alloc(wtB);
  unsigned short* qb   = (unsigned short*)alloc(actB);
  unsigned short* kbuf = (unsigned short*)alloc(actB);
  unsigned short* vrm  = (unsigned short*)alloc(actB);
  unsigned short* vtb  = (unsigned short*)alloc(actB);
  unsigned short* attb = (unsigned short*)alloc(actB);
  float*          ao   = (float*)alloc(actF);
  float*          o1f  = (float*)alloc(actF);
  unsigned short* o1b  = (unsigned short*)alloc(actB);
  unsigned short* h1   = (unsigned short*)alloc(actB);
  float*          f2   = (float*)alloc(actF);

  const int elemBlocks = (ROWS * DD) / 256;   // 16384
  const int wtBlocks   = (DD * DD) / 256;     // 1024
  dim3 gemmGrid(DD / BT, ROWS / BT);          // (4, 64)

  // 1) bf16 conversion + masking
  cvt_act_kernel<<<elemBlocks, 256, 0, stream>>>(x, xm, xb);
  cvt_act_kernel<<<elemBlocks, 256, 0, stream>>>(y, ym, yb);
  cvt_wt_kernel<<<wtBlocks, 256, 0, stream>>>(Wq, Wqt);
  cvt_wt_kernel<<<wtBlocks, 256, 0, stream>>>(Wk, Wkt);
  cvt_wt_kernel<<<wtBlocks, 256, 0, stream>>>(Wv, Wvt);
  cvt_wt_kernel<<<wtBlocks, 256, 0, stream>>>(Wo, Wot);
  cvt_wt_kernel<<<wtBlocks, 256, 0, stream>>>(W1, W1t);
  cvt_wt_kernel<<<wtBlocks, 256, 0, stream>>>(W2, W2t);

  // 2) QKV projections (WMMA + TDM-staged tiles)
  gemm_bf16_kernel<0><<<gemmGrid, 128, 0, stream>>>(xb, Wqt, nullptr, qb);
  gemm_bf16_kernel<0><<<gemmGrid, 128, 0, stream>>>(yb, Wkt, nullptr, kbuf);
  gemm_bf16_kernel<0><<<gemmGrid, 128, 0, stream>>>(yb, Wvt, nullptr, vrm);
  transpose_v_kernel<<<elemBlocks, 256, 0, stream>>>(vrm, vtb);

  // 3) attention (WMMA + LDS softmax), 128 KB LDS per block
  attn_kernel<<<BB * HH * (NN / 32), 64, 32 * MM * sizeof(float), stream>>>(
      qb, kbuf, vtb, xm, ym, attb);

  // 4) output projection + residual + LN1
  gemm_bf16_kernel<1><<<gemmGrid, 128, 0, stream>>>(attb, Wot, nullptr, ao);
  ln_res_kernel<0><<<ROWS / 8, 256, 0, stream>>>(x, ao, g1, be1, xm, o1f, o1b);

  // 5) FFN: relu(o1@W1+b1) @ W2 + b2, residual + LN2 + mask
  gemm_bf16_kernel<6><<<gemmGrid, 128, 0, stream>>>(o1b, W1t, b1, h1);
  gemm_bf16_kernel<3><<<gemmGrid, 128, 0, stream>>>(h1, W2t, b2, f2);
  ln_res_kernel<1><<<ROWS / 8, 256, 0, stream>>>(o1f, f2, g2, be2, xm, out, nullptr);
}
